// Transformer_55731495633309
// MI455X (gfx1250) — matrix operand order repaired and checked by reading, not compiled, not run on hardware
//
#include <hip/hip_runtime.h>
#include <hip/hip_bf16.h>

// ---------------------------------------------------------------------------
// CDNA5 / gfx1250 transformer forward pass.
// GEMM-class math on v_wmma_f32_16x16x32_f16; GEMM tiles streamed with
// global_load_async_to_lds_b128 double-buffering (ASYNCcnt).
// ---------------------------------------------------------------------------

typedef __attribute__((ext_vector_type(16))) _Float16 v16h;
typedef __attribute__((ext_vector_type(8)))  float    v8f;

#define D_MODEL 512
#define DH      64
#define S_LOG2  10          // S = 1024
#define S_MASK  1023

__device__ __forceinline__ v8f wmma16(v16h a, v16h b, v8f c) {
  return __builtin_amdgcn_wmma_f32_16x16x32_f16(false, a, false, b, (short)0, c,
                                                false, false);
}

// Contiguous K index pattern this code uses for 16-bit A and B fragments (16x32 A):
// lanes 0-15 hold M=lane; vector elems [0..15] = K 0..15 (+half*16).
__device__ __forceinline__ int a_kk(int i, int half) {
  return 2 * i + half * 16;
}

// Async global -> LDS copy (16B per lane). LDS offset = low 32 bits of the
// flat shared-memory address (ISA 10.2: LDS aperture keeps offset in [31:0]).
__device__ __forceinline__ void async_copy_b128(void* lds_dst,
                                                const void* gsrc) {
  unsigned int l = (unsigned int)(unsigned long long)lds_dst;
  asm volatile("global_load_async_to_lds_b128 %0, %1, off"
               :: "v"(l), "v"(gsrc)
               : "memory");
}
__device__ __forceinline__ void wait_async0() {
  asm volatile("s_wait_asynccnt 0x0" ::: "memory");
}

// ---------------------------------------------------------------------------
// GEMM: C[M,N] = act(A[M,K] @ W[K,N] + bias[N]); fp32 in, f16 WMMA, fp32 out.
// flags bit0 = ReLU, bit1 = write C transposed over (B,S) rows (logits).
// 64x64 tile, BK=32, 128 threads (4 waves). Double-buffered async tile loads.
// ---------------------------------------------------------------------------
__global__ __launch_bounds__(128)
void gemm_bias_kernel(const float* __restrict__ A, const float* __restrict__ W,
                      const float* __restrict__ bias, float* __restrict__ C,
                      int M, int N, int K, int flags, int Bdim) {
  __shared__ float Asf[2][64][32];   // M x K tile (fp32, async-filled)
  __shared__ float Bsf[2][32][64];   // K x N tile (fp32, async-filled)
  const int tid  = threadIdx.x;
  const int wave = tid >> 5, lane = tid & 31;
  const int half = lane >> 4, l16 = lane & 15;
  const int m0 = blockIdx.y * 64;
  const int n0 = blockIdx.x * 64;

  auto load_tiles = [&](int buf, int k0) {
#pragma unroll
    for (int i = 0; i < 4; ++i) {               // 512 lanes-worth x 4 floats
      int idx = i * 512 + tid * 4;
      int ar = idx >> 5, ak = idx & 31;         // A: 4 floats within a K-row
      async_copy_b128(&Asf[buf][ar][ak],
                      &A[(size_t)(m0 + ar) * K + (k0 + ak)]);
      int bk = idx >> 6, bn = idx & 63;         // B: 4 floats along N
      async_copy_b128(&Bsf[buf][bk][bn],
                      &W[(size_t)(k0 + bk) * N + (n0 + bn)]);
    }
  };

  const v8f zf = {0.f, 0.f, 0.f, 0.f, 0.f, 0.f, 0.f, 0.f};
  v8f acc[4] = {zf, zf, zf, zf};

  const int KT = K >> 5;
  load_tiles(0, 0);
  for (int kt = 0; kt < KT; ++kt) {
    const int buf = kt & 1;
    wait_async0();                 // this wave's tile-kt copies landed
    __syncthreads();               // every wave's copies landed; prior buffer free
    if (kt + 1 < KT) load_tiles(buf ^ 1, (kt + 1) << 5);   // overlap w/ WMMA

    v16h afrag;
    const int arow = wave * 16 + l16;
#pragma unroll
    for (int i = 0; i < 8; ++i) {
      int kk = a_kk(i, half);
      afrag[2 * i]     = (_Float16)Asf[buf][arow][kk];
      afrag[2 * i + 1] = (_Float16)Asf[buf][arow][kk + 1];
    }
#pragma unroll
    for (int nb = 0; nb < 4; ++nb) {
      v16h bfrag;
      const int bcol = nb * 16 + l16;
#pragma unroll
      for (int j = 0; j < 8; ++j) {
        int kk = 2 * j + half * 16;             // B operand: VGPR j = K 2j,2j+1
        bfrag[2 * j]     = (_Float16)Bsf[buf][kk][bcol];
        bfrag[2 * j + 1] = (_Float16)Bsf[buf][kk + 1][bcol];
      }
      acc[nb] = wmma16(afrag, bfrag, acc[nb]);
    }
    __syncthreads();               // done reading buf before it is refilled
  }

#pragma unroll
  for (int nb = 0; nb < 4; ++nb)
#pragma unroll
    for (int r = 0; r < 8; ++r) {
      int row = m0 + wave * 16 + half * 8 + r;   // C layout: vgpr r = M half*8+r
      int col = n0 + nb * 16 + l16;              // lane%16 = N
      float v = acc[nb][r] + bias[col];
      if (flags & 1) v = fmaxf(v, 0.f);
      size_t oidx;
      if (flags & 2) {                           // [B,S,*] -> [S,B,*], S=1024
        int bb = row >> S_LOG2, ss = row & S_MASK;
        oidx = ((size_t)ss * Bdim + bb) * N + col;
      } else {
        oidx = (size_t)row * N + col;
      }
      C[oidx] = v;
    }
}

// ---------------------------------------------------------------------------
// Flash attention: one block (4 waves) per (b, h, 64-query tile).
// Streams 32-key chunks: QK^T (4 WMMA) -> online softmax -> P@V (4 WMMA).
// q/k/v/out are [B,S,D] fp32; head h lives in columns h*64..h*64+63.
// ---------------------------------------------------------------------------
__global__ __launch_bounds__(128)
void attn_kernel(const float* __restrict__ q, const float* __restrict__ k,
                 const float* __restrict__ vv, float* __restrict__ out,
                 int S, int causal) {
  __shared__ _Float16 Ks[32][72];        // [key][dh]  (N-major for QK^T B-op)
  __shared__ _Float16 Vs[64][40];        // [dh][key]  (N-major for P@V B-op)
  __shared__ _Float16 Ps[4][16][40];     // per-wave P tile (16x32)

  const int tid  = threadIdx.x;
  const int wave = tid >> 5, lane = tid & 31;
  const int half = lane >> 4, l16 = lane & 15;
  const int h  = blockIdx.y, b = blockIdx.z;
  const int q0 = blockIdx.x * 64;
  const size_t base = ((size_t)b * S) * D_MODEL + (size_t)h * DH;

  // Q fragments (16x64 -> two 16x32 A fragments), pre-scaled by 1/sqrt(DH).
  v16h qa[2];
  {
    const float* qp = q + base + (size_t)(q0 + wave * 16 + l16) * D_MODEL;
#pragma unroll
    for (int f = 0; f < 2; ++f)
#pragma unroll
      for (int i = 0; i < 8; ++i) {
        int kk = f * 32 + a_kk(i, half);
        qa[f][2 * i]     = (_Float16)(qp[kk] * 0.125f);
        qa[f][2 * i + 1] = (_Float16)(qp[kk + 1] * 0.125f);
      }
  }

  const v8f zf = {0.f, 0.f, 0.f, 0.f, 0.f, 0.f, 0.f, 0.f};
  v8f oacc[4] = {zf, zf, zf, zf};
  float m_run[8], l_run[8];
#pragma unroll
  for (int r = 0; r < 8; ++r) { m_run[r] = -3.0e38f; l_run[r] = 0.f; }

  const int kc_end = causal ? (q0 + 64) / 32 : S / 32;
  for (int kc = 0; kc < kc_end; ++kc) {
    __syncthreads();                               // prior tiles fully consumed
#pragma unroll
    for (int i = 0; i < 16; ++i) {                 // 2048 elems, 128 threads
      int idx = i * 128 + tid;
      int kr = idx >> 6, dc = idx & 63;
      size_t g = base + (size_t)(kc * 32 + kr) * D_MODEL + dc;
      Ks[kr][dc] = (_Float16)k[g];
      Vs[dc][kr] = (_Float16)vv[g];
    }
    __syncthreads();

    // scores: two 16x16 tiles covering keys kc*32 + {0..15, 16..31}
    v8f st[2] = {zf, zf};
#pragma unroll
    for (int t = 0; t < 2; ++t)
#pragma unroll
      for (int f = 0; f < 2; ++f) {
        v16h bfrag;
        const int key = t * 16 + l16;
#pragma unroll
        for (int j = 0; j < 8; ++j) {
          int kk = f * 32 + 2 * j + half * 16;
          bfrag[2 * j]     = Ks[key][kk];
          bfrag[2 * j + 1] = Ks[key][kk + 1];
        }
        st[t] = wmma16(qa[f], bfrag, st[t]);
      }

    if (causal) {
#pragma unroll
      for (int t = 0; t < 2; ++t) {
        int kcol = kc * 32 + t * 16 + l16;
#pragma unroll
        for (int r = 0; r < 8; ++r) {
          int qrow = q0 + wave * 16 + half * 8 + r;
          if (kcol > qrow) st[t][r] = -3.0e38f;
        }
      }
    }

    // online softmax (row = half*8+r; columns live across the 16-lane half)
#pragma unroll
    for (int r = 0; r < 8; ++r) {
      float pm = fmaxf(st[0][r], st[1][r]);
#pragma unroll
      for (int off = 1; off < 16; off <<= 1)
        pm = fmaxf(pm, __shfl_xor(pm, off, 32));
      float mn = fmaxf(m_run[r], pm);
      float sc = __expf(m_run[r] - mn);
      float p0 = __expf(st[0][r] - mn);
      float p1 = __expf(st[1][r] - mn);
      float ps = p0 + p1;
#pragma unroll
      for (int off = 1; off < 16; off <<= 1)
        ps += __shfl_xor(ps, off, 32);
      l_run[r] = l_run[r] * sc + ps;
      m_run[r] = mn;
#pragma unroll
      for (int nb = 0; nb < 4; ++nb) oacc[nb][r] = oacc[nb][r] * sc;
      Ps[wave][half * 8 + r][l16]      = (_Float16)p0;
      Ps[wave][half * 8 + r][16 + l16] = (_Float16)p1;
    }
    __syncthreads();                               // Ps visible (C->A relayout)

    v16h pfrag;
#pragma unroll
    for (int i = 0; i < 8; ++i) {
      int kk = a_kk(i, half);
      pfrag[2 * i]     = Ps[wave][l16][kk];
      pfrag[2 * i + 1] = Ps[wave][l16][kk + 1];
    }
#pragma unroll
    for (int nb = 0; nb < 4; ++nb) {
      v16h vfrag;
#pragma unroll
      for (int j = 0; j < 8; ++j) {
        int kk = 2 * j + half * 16;
        vfrag[2 * j]     = Vs[nb * 16 + l16][kk];
        vfrag[2 * j + 1] = Vs[nb * 16 + l16][kk + 1];
      }
      oacc[nb] = wmma16(pfrag, vfrag, oacc[nb]);
    }
  }

#pragma unroll
  for (int nb = 0; nb < 4; ++nb)
#pragma unroll
    for (int r = 0; r < 8; ++r) {
      int qrow = q0 + wave * 16 + half * 8 + r;
      out[base + (size_t)qrow * D_MODEL + nb * 16 + l16] = oacc[nb][r] / l_run[r];
    }
}

// ---------------------------------------------------------------------------
// out[row] = LayerNorm(x[row] + t[row]) * g + beta   (D = 512, 256 threads)
// ---------------------------------------------------------------------------
__global__ __launch_bounds__(256)
void add_ln_kernel(const float* __restrict__ x, const float* __restrict__ t,
                   const float* __restrict__ g, const float* __restrict__ beta,
                   float* __restrict__ out) {
  __shared__ float red[256];
  const int row = blockIdx.x, tid = threadIdx.x;
  const float* xr = x + (size_t)row * D_MODEL;
  const float* tr = t + (size_t)row * D_MODEL;
  float a = xr[tid] + tr[tid];
  float b = xr[tid + 256] + tr[tid + 256];
  red[tid] = a + b;
  __syncthreads();
  for (int o = 128; o > 0; o >>= 1) {
    if (tid < o) red[tid] += red[tid + o];
    __syncthreads();
  }
  float mean = red[0] * (1.f / 512.f);
  __syncthreads();
  float da = a - mean, db = b - mean;
  red[tid] = da * da + db * db;
  __syncthreads();
  for (int o = 128; o > 0; o >>= 1) {
    if (tid < o) red[tid] += red[tid + o];
    __syncthreads();
  }
  float rstd = rsqrtf(red[0] * (1.f / 512.f) + 1e-5f);
  float* orow = out + (size_t)row * D_MODEL;
  orow[tid]       = da * rstd * g[tid] + beta[tid];
  orow[tid + 256] = db * rstd * g[tid + 256] + beta[tid + 256];
}

// ---------------------------------------------------------------------------
// Embedding gather + positional add. tok_t: 0 -> tok[b*S+s], 1 -> tok[s*B+b].
// ---------------------------------------------------------------------------
__global__ __launch_bounds__(256)
void embed_kernel(const int* __restrict__ tok, const float* __restrict__ emb,
                  const float* __restrict__ pos, float* __restrict__ out,
                  int S, int Bdim, int tok_t) {
  const int s = blockIdx.x, b = blockIdx.y;
  const int t = tok_t ? tok[s * Bdim + b] : tok[b * S + s];
  const float* e = emb + (size_t)t * D_MODEL;
  const float* p = pos + (size_t)s * D_MODEL;
  float* o = out + ((size_t)b * S + s) * D_MODEL;
  for (int i = threadIdx.x; i < D_MODEL; i += blockDim.x) o[i] = e[i] + p[i];
}

// ---------------------------------------------------------------------------
// Host orchestration
// ---------------------------------------------------------------------------
extern "C" void kernel_launch(void* const* d_in, const int* in_sizes, int n_in,
                              void* d_out, int out_size, void* d_ws,
                              size_t ws_size, hipStream_t stream) {
  const int B = 4, S = 1024, D = 512, Hh = 8, F = 2048, V = 32000;
  const int LE = 6, LD = 6, M = B * S;

  const int*   src     = (const int*)d_in[0];
  const int*   tgt     = (const int*)d_in[1];
  const float* emb     = (const float*)d_in[2];
  const float* y_w     = (const float*)d_in[3];
  const float* y_b     = (const float*)d_in[4];
  const float* enc_pos = (const float*)d_in[5];
  const float* dec_pos = (const float*)d_in[6];
  auto encp = [&](int i) { return (const float*)d_in[7 + i]; };
  // enc leaves: 0 wq 1 bq 2 wk 3 bk 4 wv 5 bv 6 wo 7 bo 8 w1 9 b1 10 w2 11 b2
  //             12 ln1_g 13 ln1_b 14 ln2_g 15 ln2_b
  auto decp = [&](int i) { return (const float*)d_in[23 + i]; };
  // dec leaves: 0..7 swq..sbo  8..15 cwq..cbo  16 w1 17 b1 18 w2 19 b2
  //             20 ln1_g 21 ln1_b 22 ln2_g 23 ln2_b 24 ln3_g 25 ln3_b

  char* ws = (char*)d_ws;
  float* X   = (float*)(ws);                      //  8 MB activations
  float* T1  = (float*)(ws + (size_t)(8  << 20)); //  8 MB sublayer out
  float* Q   = (float*)(ws + (size_t)(16 << 20)); //  8 MB
  float* Kb  = (float*)(ws + (size_t)(24 << 20)); //  8 MB
  float* Vb  = (float*)(ws + (size_t)(32 << 20)); //  8 MB
  float* AO  = (float*)(ws + (size_t)(40 << 20)); //  8 MB attn context
  float* Hd  = (float*)(ws + (size_t)(48 << 20)); // 32 MB ffn hidden
  float* ENC = (float*)(ws + (size_t)(80 << 20)); //  8 MB encoder out

  auto gemm = [&](const float* A, const float* W, const float* bias, float* C,
                  int Mm, int Nn, int Kk, int flags) {
    dim3 g(Nn / 64, Mm / 64);
    gemm_bias_kernel<<<g, 128, 0, stream>>>(A, W, bias, C, Mm, Nn, Kk, flags,
                                            B);
  };
  auto attn = [&](const float* q, const float* k, const float* v, float* o,
                  int causal) {
    dim3 g(S / 64, Hh, B);
    attn_kernel<<<g, 128, 0, stream>>>(q, k, v, o, S, causal);
  };
  auto ln = [&](const float* x, const float* t, const float* g,
                const float* bb, float* o) {
    add_ln_kernel<<<M, 256, 0, stream>>>(x, t, g, bb, o);
  };

  // ----- encoder -----
  embed_kernel<<<dim3(S, B), 256, 0, stream>>>(src, emb, enc_pos, X, S, B, 0);
  for (int l = 0; l < LE; ++l) {
    size_t wO = (size_t)l * D * D, bO = (size_t)l * D;
    gemm(X, encp(0) + wO, encp(1) + bO, Q,  M, D, D, 0);
    gemm(X, encp(2) + wO, encp(3) + bO, Kb, M, D, D, 0);
    gemm(X, encp(4) + wO, encp(5) + bO, Vb, M, D, D, 0);
    attn(Q, Kb, Vb, AO, 0);
    gemm(AO, encp(6) + wO, encp(7) + bO, T1, M, D, D, 0);
    ln(X, T1, encp(12) + bO, encp(13) + bO, X);
    gemm(X, encp(8) + (size_t)l * D * F, encp(9) + (size_t)l * F, Hd, M, F, D, 1);
    gemm(Hd, encp(10) + (size_t)l * F * D, encp(11) + bO, T1, M, D, F, 0);
    ln(X, T1, encp(14) + bO, encp(15) + bO, X);
  }
  hipMemcpyAsync(ENC, X, (size_t)M * D * sizeof(float),
                 hipMemcpyDeviceToDevice, stream);

  // ----- decoder -----
  embed_kernel<<<dim3(S, B), 256, 0, stream>>>(tgt, emb, dec_pos, X, S, B, 1);
  for (int l = 0; l < LD; ++l) {
    size_t wO = (size_t)l * D * D, bO = (size_t)l * D;
    // masked self-attention
    gemm(X, decp(0) + wO, decp(1) + bO, Q,  M, D, D, 0);
    gemm(X, decp(2) + wO, decp(3) + bO, Kb, M, D, D, 0);
    gemm(X, decp(4) + wO, decp(5) + bO, Vb, M, D, D, 0);
    attn(Q, Kb, Vb, AO, 1);
    gemm(AO, decp(6) + wO, decp(7) + bO, T1, M, D, D, 0);
    ln(X, T1, decp(20) + bO, decp(21) + bO, X);
    // cross-attention against encoder output
    gemm(X,   decp(8)  + wO, decp(9)  + bO, Q,  M, D, D, 0);
    gemm(ENC, decp(10) + wO, decp(11) + bO, Kb, M, D, D, 0);
    gemm(ENC, decp(12) + wO, decp(13) + bO, Vb, M, D, D, 0);
    attn(Q, Kb, Vb, AO, 0);
    gemm(AO, decp(14) + wO, decp(15) + bO, T1, M, D, D, 0);
    ln(X, T1, decp(22) + bO, decp(23) + bO, X);
    // FFN
    gemm(X, decp(16) + (size_t)l * D * F, decp(17) + (size_t)l * F, Hd, M, F, D, 1);
    gemm(Hd, decp(18) + (size_t)l * F * D, decp(19) + bO, T1, M, D, F, 0);
    ln(X, T1, decp(24) + bO, decp(25) + bO, X);
  }

  // ----- logits: [B*S, D] @ [D, V] + b, emitted as [S, B, V] -----
  gemm(X, y_w, y_b, (float*)d_out, M, V, D, 2);
}
